// Block_63221918597210
// MI455X (gfx1250) — matrix operand order repaired and checked by reading, not compiled, not run on hardware
//
#include <hip/hip_runtime.h>

typedef __bf16 bf16;
typedef __bf16 v16bf __attribute__((ext_vector_type(16)));
typedef float  v8f   __attribute__((ext_vector_type(8)));

#define D_INNER 2048
#define N_EMBED 1024
#define D_STATE 16
#define DT_RANK 64
#define D_FF    4096
#define SEQ_L   2048
#define NBATCH  2
#define MROWS   4096   // NBATCH * SEQ_L

__device__ __forceinline__ float sigmoidf_(float x){ return 1.0f/(1.0f + __expf(-x)); }

// ---------------- fp32 -> bf16 convert ----------------
__global__ void cvt_bf16_kernel(const float* __restrict__ src, bf16* __restrict__ dst, int n){
  int i = blockIdx.x*256 + threadIdx.x;
  if (i < n) dst[i] = (bf16)src[i];
}

// ---------------- layernorm (fp32 in) -> bf16 out ----------------
__global__ __launch_bounds__(256)
void ln_bf16_kernel(const float* __restrict__ x, const float* __restrict__ w,
                    const float* __restrict__ b, bf16* __restrict__ out, int C){
  int row = blockIdx.x;
  const float* xr = x + (size_t)row*C;
  float s = 0.f, s2 = 0.f;
  for (int c = threadIdx.x; c < C; c += 256){ float v = xr[c]; s += v; s2 += v*v; }
  __shared__ float r0[256], r1[256];
  r0[threadIdx.x] = s; r1[threadIdx.x] = s2;
  __syncthreads();
  for (int off = 128; off > 0; off >>= 1){
    if (threadIdx.x < off){
      r0[threadIdx.x] += r0[threadIdx.x + off];
      r1[threadIdx.x] += r1[threadIdx.x + off];
    }
    __syncthreads();
  }
  float mean = r0[0] / C;
  float var  = r1[0] / C - mean*mean;
  float rstd = rsqrtf(var + 1e-5f);
  for (int c = threadIdx.x; c < C; c += 256)
    out[(size_t)row*C + c] = (bf16)((xr[c]-mean)*rstd*w[c] + b[c]);
}

// ---------------- WMMA bf16 GEMM: C[M,N] = A[M,K] * W[N,K]^T (+epilogue) ----------------
// 128x128x32 block tile, 8 waves (wave32), wave tile 32x64 (2x4 WMMA tiles),
// software-pipelined global->reg->LDS with ping-pong LDS buffers (1 barrier / K-step).
enum { EPI_F32 = 0, EPI_BIAS_SOFTPLUS = 1, EPI_RES = 2, EPI_BIAS_RELU_BF16 = 3, EPI_BIAS_RES = 4 };

template<int EPI>
__global__ __launch_bounds__(256)
void gemm_bf16_wmma(const bf16* __restrict__ A, const bf16* __restrict__ W,
                    int M, int N, int K,
                    const float* __restrict__ bias, const float* __restrict__ res,
                    float* __restrict__ outf, bf16* __restrict__ outb){
  constexpr int BM = 128, BN = 128, BK = 32, LDT = 40;   // LDT halves (80B rows, LDS pad)
  __shared__ bf16 As[2*BM*LDT];
  __shared__ bf16 Bs[2*BN*LDT];

  const int tid  = threadIdx.x;
  const int wid  = tid >> 5;        // wave32
  const int lane = tid & 31;
  const int wm   = wid & 3;         // 0..3 : 32 rows each
  const int wn   = wid >> 2;        // 0..1 : 64 cols each
  const int half = lane >> 4;
  const int mr   = lane & 15;
  const int m0   = blockIdx.y * BM;
  const int n0   = blockIdx.x * BN;

  // staging chunk coordinates (2 chunks for A, 2 for B per thread)
  const int rA0 = (tid      ) >> 2, cA0 = (tid      ) & 3;
  const int rA1 = (tid + 256) >> 2, cA1 = (tid + 256) & 3;

  v8f acc[2][4];
#pragma unroll
  for (int i = 0; i < 2; ++i)
#pragma unroll
    for (int j = 0; j < 4; ++j) acc[i][j] = {};

  const int ktiles = K / BK;
  float4 ga0, ga1, gb0, gb1;

  auto load_tile = [&](int kt){
    const int k0 = kt * BK;
    ga0 = *reinterpret_cast<const float4*>(A + (size_t)(m0+rA0)*K + k0 + cA0*8);
    ga1 = *reinterpret_cast<const float4*>(A + (size_t)(m0+rA1)*K + k0 + cA1*8);
    // branchless clamped B loads, zeroed if row OOB (avoids exec-masked serialization)
    int nb0 = n0 + rA0, nb1 = n0 + rA1;
    int sb0 = (nb0 < N) ? nb0 : 0;
    int sb1 = (nb1 < N) ? nb1 : 0;
    gb0 = *reinterpret_cast<const float4*>(W + (size_t)sb0*K + k0 + cA0*8);
    gb1 = *reinterpret_cast<const float4*>(W + (size_t)sb1*K + k0 + cA1*8);
    if (nb0 >= N){ gb0.x = 0.f; gb0.y = 0.f; gb0.z = 0.f; gb0.w = 0.f; }
    if (nb1 >= N){ gb1.x = 0.f; gb1.y = 0.f; gb1.z = 0.f; gb1.w = 0.f; }
  };

  load_tile(0);

  for (int kt = 0; kt < ktiles; ++kt){
    const int p = kt & 1;
    bf16* Ab = As + p*(BM*LDT);
    bf16* Bb = Bs + p*(BN*LDT);

    // commit staged registers to this iteration's LDS buffer
    *reinterpret_cast<float4*>(&Ab[rA0*LDT + cA0*8]) = ga0;
    *reinterpret_cast<float4*>(&Ab[rA1*LDT + cA1*8]) = ga1;
    *reinterpret_cast<float4*>(&Bb[rA0*LDT + cA0*8]) = gb0;
    *reinterpret_cast<float4*>(&Bb[rA1*LDT + cA1*8]) = gb1;
    __syncthreads();

    // issue next tile's global loads while WMMAs consume this tile
    if (kt + 1 < ktiles) load_tile(kt + 1);
    if (kt + 2 < ktiles)
      __builtin_prefetch(A + (size_t)(m0 + rA0)*K + (kt+2)*BK, 0, 1);

    // ---- fragments from LDS (ds_load_b128) ----
    v16bf afr[2], bfr[4];
#pragma unroll
    for (int i = 0; i < 2; ++i){
      int row = wm*32 + i*16 + mr;
      float4* fp = reinterpret_cast<float4*>(&afr[i]);
      // A 16x32 bf16 layout: lane half selects K chunks {8h..8h+7, 16+8h..23+8h}
      fp[0] = *reinterpret_cast<const float4*>(&Ab[row*LDT + 8*half]);
      fp[1] = *reinterpret_cast<const float4*>(&Ab[row*LDT + 16 + 8*half]);
    }
#pragma unroll
    for (int j = 0; j < 4; ++j){
      int row = wn*64 + j*16 + mr;   // row of W == output column
      float4* fp = reinterpret_cast<float4*>(&bfr[j]);
      // B 32x16 bf16 layout: lane half selects K chunks {8h..8h+7, 16+8h..23+8h}
      fp[0] = *reinterpret_cast<const float4*>(&Bb[row*LDT + 8*half]);
      fp[1] = *reinterpret_cast<const float4*>(&Bb[row*LDT + 16 + 8*half]);
    }

#pragma unroll
    for (int i = 0; i < 2; ++i)
#pragma unroll
      for (int j = 0; j < 4; ++j)
        acc[i][j] = __builtin_amdgcn_wmma_f32_16x16x32_bf16(
            false, afr[i], false, bfr[j], (short)0, acc[i][j], false, false);
    // no trailing barrier: ping-pong buffer + the single barrier above give
    // max inter-wave skew < 1 iteration, so buffer p is only overwritten
    // after every wave has passed the barrier that precedes its reuse.
  }

  // ---- epilogue ----
#pragma unroll
  for (int i = 0; i < 2; ++i)
#pragma unroll
    for (int j = 0; j < 4; ++j){
#pragma unroll
      for (int r = 0; r < 8; ++r){
        int row = m0 + wm*32 + i*16 + half*8 + r;
        int col = n0 + wn*64 + j*16 + mr;
        if (col >= N) continue;
        size_t idx = (size_t)row*N + col;
        float v = acc[i][j][r];
        if constexpr (EPI == EPI_BIAS_SOFTPLUS){
          v += bias[col];
          v = (v > 20.f) ? v : log1pf(__expf(v));
          outf[idx] = v;
        } else if constexpr (EPI == EPI_RES){
          outf[idx] = v + res[idx];
        } else if constexpr (EPI == EPI_BIAS_RELU_BF16){
          v = fmaxf(v + bias[col], 0.f);
          outb[idx] = (bf16)v;
        } else if constexpr (EPI == EPI_BIAS_RES){
          outf[idx] = v + bias[col] + res[idx];
        } else {
          outf[idx] = v;
        }
      }
    }
}

// ---------------- depthwise causal conv (D_CONV=4) + SiLU -> bf16 ----------------
// strip-mined: each thread owns one channel for 128 timesteps, sliding window in regs
__global__ __launch_bounds__(256)
void conv_silu_kernel(const float* __restrict__ xz, const float* __restrict__ cw,
                      const float* __restrict__ cb, bf16* __restrict__ xcb, int L){
  const int d  = blockIdx.x*256 + threadIdx.x;   // < D_INNER
  const int b  = blockIdx.z;
  const int l0 = blockIdx.y * 128;
  const float w0 = cw[d*4+0], w1 = cw[d*4+1], w2 = cw[d*4+2], w3 = cw[d*4+3];
  const float bia = cb[d];
  float xm3 = 0.f, xm2 = 0.f, xm1 = 0.f;
  if (l0 - 3 >= 0) xm3 = xz[(size_t)(b*L + l0-3)*(2*D_INNER) + d];
  if (l0 - 2 >= 0) xm2 = xz[(size_t)(b*L + l0-2)*(2*D_INNER) + d];
  if (l0 - 1 >= 0) xm1 = xz[(size_t)(b*L + l0-1)*(2*D_INNER) + d];
  for (int l = l0; l < l0 + 128; ++l){
    float xc  = xz[(size_t)(b*L + l)*(2*D_INNER) + d];
    float acc = bia + w0*xm3 + w1*xm2 + w2*xm1 + w3*xc;
    acc = acc * sigmoidf_(acc);
    xcb[(size_t)(b*L + l)*D_INNER + d] = (bf16)acc;
    xm3 = xm2; xm2 = xm1; xm1 = xc;
  }
}

// ---------------- slice dt columns of xdb -> bf16 ----------------
__global__ void slice_dt_kernel(const float* __restrict__ xdb, bf16* __restrict__ dtin){
  int i = blockIdx.x*256 + threadIdx.x;   // MROWS*DT_RANK
  int r = i >> 6, c = i & 63;
  dtin[i] = (bf16)xdb[r*(DT_RANK + 2*D_STATE) + c];
}

// ---------------- selective scan, fused D-skip + SiLU(z) gate -> bf16 ----------------
__global__ __launch_bounds__(256)
void scan_kernel(const float* __restrict__ xz, const bf16* __restrict__ xcb,
                 const float* __restrict__ dt, const float* __restrict__ xdb,
                 const float* __restrict__ A_log, const float* __restrict__ Dskip,
                 bf16* __restrict__ ybf, int L){
  const int b = blockIdx.y;
  const int d = blockIdx.x*256 + threadIdx.x;   // channel in [0, D_INNER)
  float Aneg[D_STATE];
#pragma unroll
  for (int s = 0; s < D_STATE; ++s) Aneg[s] = -__expf(A_log[d*D_STATE + s]);
  const float Dsk = Dskip[d];
  float h[D_STATE];
#pragma unroll
  for (int s = 0; s < D_STATE; ++s) h[s] = 0.f;

  __shared__ float BC[2*D_STATE];
  for (int l = 0; l < L; ++l){
    const size_t row = (size_t)b*L + l;
    if (threadIdx.x < 2*D_STATE)
      BC[threadIdx.x] = xdb[row*(DT_RANK + 2*D_STATE) + DT_RANK + threadIdx.x];
    __syncthreads();
    const float dtv = dt[row*D_INNER + d];
    const float xv  = (float)xcb[row*D_INNER + d];
    float y = 0.f;
#pragma unroll
    for (int s = 0; s < D_STATE; ++s){
      float dA = __expf(dtv * Aneg[s]);
      h[s] = dA*h[s] + dtv * BC[s] * xv;
      y += h[s] * BC[D_STATE + s];
    }
    y += xv * Dsk;
    const float z = xz[row*(2*D_INNER) + D_INNER + d];
    y *= z * sigmoidf_(z);
    ybf[row*D_INNER + d] = (bf16)y;
    __syncthreads();
  }
}

extern "C" void kernel_launch(void* const* d_in, const int* in_sizes, int n_in,
                              void* d_out, int out_size, void* d_ws, size_t ws_size,
                              hipStream_t stream) {
  (void)in_sizes; (void)n_in; (void)out_size; (void)ws_size;
  const float* x         = (const float*)d_in[0];
  const float* ln1_w     = (const float*)d_in[1];
  const float* ln1_b     = (const float*)d_in[2];
  const float* ln2_w     = (const float*)d_in[3];
  const float* ln2_b     = (const float*)d_in[4];
  const float* in_proj_w = (const float*)d_in[5];
  const float* conv_w    = (const float*)d_in[6];
  const float* conv_b    = (const float*)d_in[7];
  const float* x_proj_w  = (const float*)d_in[8];
  const float* dt_proj_w = (const float*)d_in[9];
  const float* dt_proj_b = (const float*)d_in[10];
  const float* A_log     = (const float*)d_in[11];
  const float* D_skip    = (const float*)d_in[12];
  const float* out_proj_w= (const float*)d_in[13];
  const float* ffn_w1    = (const float*)d_in[14];
  const float* ffn_b1    = (const float*)d_in[15];
  const float* ffn_w2    = (const float*)d_in[16];
  const float* ffn_b2    = (const float*)d_in[17];

  char* ws = (char*)d_ws;
  bf16*  W_IN = (bf16*)(ws + 0);           // 4096x1024 bf16
  bf16*  W_XP = (bf16*)(ws + 8388608);     // 96x2048
  bf16*  W_DT = (bf16*)(ws + 8781824);     // 2048x64
  bf16*  W_OUT= (bf16*)(ws + 9043968);     // 1024x2048
  bf16*  W_F1 = (bf16*)(ws + 13238272);    // 4096x1024
  bf16*  W_F2 = (bf16*)(ws + 21626880);    // 1024x4096
  bf16*  XN   = (bf16*)(ws + 30015488);    // 4096x1024 bf16 (later reused as H)
  float* XZ   = (float*)(ws + 38404096);   // 4096x4096 f32  (later reused as R1 bf16)
  bf16*  XCB  = (bf16*)(ws + 105512960);   // 4096x2048 bf16 (later reused as X2 f32)
  float* XDB  = (float*)(ws + 122290176);  // 4096x96 f32
  bf16*  DTIN = (bf16*)(ws + 123863040);   // 4096x64 bf16
  float* DT   = (float*)(ws + 124387328);  // 4096x2048 f32
  bf16*  Y    = (bf16*)(ws + 157941760);   // 4096x2048 bf16
  float* X2   = (float*)XCB;               // alias: xc dead after scan
  bf16*  R1   = (bf16*)XZ;                 // alias: xz dead after scan
  bf16*  H    = XN;                        // alias: xn dead after in_proj GEMM

  auto cvt = [&](const float* s, bf16* dst, int n){
    cvt_bf16_kernel<<<(n + 255)/256, 256, 0, stream>>>(s, dst, n);
  };
  cvt(in_proj_w,  W_IN,  2*D_INNER*N_EMBED);
  cvt(x_proj_w,   W_XP,  (DT_RANK + 2*D_STATE)*D_INNER);
  cvt(dt_proj_w,  W_DT,  D_INNER*DT_RANK);
  cvt(out_proj_w, W_OUT, N_EMBED*D_INNER);
  cvt(ffn_w1,     W_F1,  D_FF*N_EMBED);
  cvt(ffn_w2,     W_F2,  N_EMBED*D_FF);

  // ln1
  ln_bf16_kernel<<<MROWS, 256, 0, stream>>>(x, ln1_w, ln1_b, XN, N_EMBED);
  // xz = xn @ in_proj_w^T  [4096 x 4096]
  gemm_bf16_wmma<EPI_F32><<<dim3(32, 32), 256, 0, stream>>>(
      XN, W_IN, MROWS, 2*D_INNER, N_EMBED, nullptr, nullptr, XZ, nullptr);
  // depthwise causal conv + SiLU
  conv_silu_kernel<<<dim3(D_INNER/256, SEQ_L/128, NBATCH), 256, 0, stream>>>(
      XZ, conv_w, conv_b, XCB, SEQ_L);
  // xdb = xc @ x_proj_w^T  [4096 x 96]
  gemm_bf16_wmma<EPI_F32><<<dim3(1, 32), 256, 0, stream>>>(
      XCB, W_XP, MROWS, DT_RANK + 2*D_STATE, D_INNER, nullptr, nullptr, XDB, nullptr);
  // dt slice -> bf16
  slice_dt_kernel<<<(MROWS*DT_RANK)/256, 256, 0, stream>>>(XDB, DTIN);
  // dt = softplus(dtin @ dt_proj_w^T + b)  [4096 x 2048]
  gemm_bf16_wmma<EPI_BIAS_SOFTPLUS><<<dim3(16, 32), 256, 0, stream>>>(
      DTIN, W_DT, MROWS, D_INNER, DT_RANK, dt_proj_b, nullptr, DT, nullptr);
  // selective scan + D-skip + SiLU(z) gate
  scan_kernel<<<dim3(D_INNER/256, NBATCH), 256, 0, stream>>>(
      XZ, XCB, DT, XDB, A_log, D_skip, Y, SEQ_L);
  // x2 = y @ out_proj_w^T + x  [4096 x 1024]
  gemm_bf16_wmma<EPI_RES><<<dim3(8, 32), 256, 0, stream>>>(
      Y, W_OUT, MROWS, N_EMBED, D_INNER, nullptr, x, X2, nullptr);
  // ln2
  ln_bf16_kernel<<<MROWS, 256, 0, stream>>>(X2, ln2_w, ln2_b, H, N_EMBED);
  // r1 = relu(h @ ffn_w1^T + b1)  [4096 x 4096] bf16
  gemm_bf16_wmma<EPI_BIAS_RELU_BF16><<<dim3(32, 32), 256, 0, stream>>>(
      H, W_F1, MROWS, D_FF, N_EMBED, ffn_b1, nullptr, nullptr, R1);
  // out = r1 @ ffn_w2^T + b2 + x2  [4096 x 1024]
  gemm_bf16_wmma<EPI_BIAS_RES><<<dim3(8, 32), 256, 0, stream>>>(
      R1, W_F2, MROWS, N_EMBED, D_FF, ffn_b2, X2, (float*)d_out, nullptr);
}
